// SelfAttention_8589935387
// MI455X (gfx1250) — matrix operand order repaired and checked by reading, not compiled, not run on hardware
//
#include <hip/hip_runtime.h>
#include <hip/hip_bf16.h>
#include <stdint.h>

typedef _Float16 f16;
typedef __attribute__((ext_vector_type(16))) _Float16 v16h;
typedef __attribute__((ext_vector_type(8)))  _Float16 v8h;
typedef __attribute__((ext_vector_type(8)))  float    v8f;

#define B_  32
#define N_  512
#define D_  768
#define H_  12
#define E_  64
#define MT_ 32   // query rows per workgroup

// ---- WMMA fragment helpers (wave32, 16x16x32 f16 -> f32) -------------------
// A-matrix 16x32 f16: lane (g=lane>>4, m=lane&15) holds row m,
//   K = [8g..8g+7] in VGPR0-3 and [16+8g..16+8g+7] in VGPR4-7.
__device__ __forceinline__ v16h frag_a(const f16* A, int lda, int lane) {
    const int g = lane >> 4, m = lane & 15;
    const f16* row = A + m * lda;
    v8h lo = *(const v8h*)(row + 8 * g);
    v8h hi = *(const v8h*)(row + 16 + 8 * g);
    v16h r;
#pragma unroll
    for (int i = 0; i < 8; ++i) { r[i] = lo[i]; r[8 + i] = hi[i]; }
    return r;
}

// B-matrix 32x16 f16 given B^T row-major (BT[n][k]): lane (g,n) holds
// column n, K = [8g..8g+7] then [16+8g..16+8g+7] in BT row n.
__device__ __forceinline__ v16h frag_bt(const f16* BT, int ldbt, int lane) {
    const int g = lane >> 4, n = lane & 15;
    const f16* p = BT + n * ldbt + 8 * g;
    v8h lo = *(const v8h*)(p);
    v8h hi = *(const v8h*)(p + 16);
    v16h r;
#pragma unroll
    for (int i = 0; i < 8; ++i) { r[i] = lo[i]; r[8 + i] = hi[i]; }
    return r;
}

__device__ __forceinline__ v8f wmma16(v16h a, v16h b, v8f c) {
    return __builtin_amdgcn_wmma_f32_16x16x32_f16(
        false, a, false, b, (short)0, c, false, false);
}

// ---- gfx1250 async global->LDS copy (ASYNCcnt-tracked) ---------------------
// Generic shared pointers carry the LDS byte offset in their low 32 bits
// (flat->LDS truncation, ISA 10.2), so we can feed them to VDST directly.
__device__ __forceinline__ void async_b128(const void* gsrc, void* ldst) {
    uint32_t l = (uint32_t)(uintptr_t)ldst;
    uint64_t ga = (uint64_t)(uintptr_t)gsrc;
    asm volatile("global_load_async_to_lds_b128 %0, %1, off"
                 :: "v"(l), "v"(ga) : "memory");
}
__device__ __forceinline__ void wait_async0() {
    asm volatile("s_wait_asynccnt 0" ::: "memory");
}

// ---- Conversion / layout kernels -------------------------------------------
__global__ void cvt_x_kernel(const float* __restrict__ x, f16* __restrict__ xh, int n) {
    int i = blockIdx.x * blockDim.x + threadIdx.x;
    if (i < n) xh[i] = (f16)x[i];
}

// Q[h][d][e] (fp32) -> Qt[h][e][d] (f16)   (B^T layout for phase 1)
__global__ void cvt_qt_kernel(const float* __restrict__ Q, f16* __restrict__ Qt) {
    int i = blockIdx.x * blockDim.x + threadIdx.x;
    if (i >= H_ * D_ * D_) return;
    int h = i / (D_ * D_);
    int r = i % (D_ * D_);
    int d = r / D_;
    int e = r % D_;
    Qt[(size_t)h * D_ * D_ + (size_t)e * D_ + d] = (f16)Q[i];
}

// V[h][d][e] (fp32, e<64) -> Vt[h][e][d] (f16)
__global__ void cvt_vt_kernel(const float* __restrict__ V, f16* __restrict__ Vt) {
    int i = blockIdx.x * blockDim.x + threadIdx.x;
    if (i >= H_ * D_ * E_) return;
    int h = i / (D_ * E_);
    int r = i % (D_ * E_);
    int d = r / E_;
    int e = r % E_;
    Vt[((size_t)h * E_ + e) * D_ + d] = (f16)V[i];
}

// ---- xv^T[b][h][e][m] = sum_d x[b][m][d] * V[h][d][e]  (WMMA GEMM) ---------
__global__ __launch_bounds__(256)
void xv_kernel(const f16* __restrict__ xh, const f16* __restrict__ Vt,
               f16* __restrict__ xvt) {
    const int tid = threadIdx.x, wave = tid >> 5, lane = tid & 31;
    const int g = lane >> 4, nn = lane & 15;
    int t = blockIdx.x * 8 + wave;           // 49152 tiles
    const int mblk = t & 31;                 // 512/16
    const int eblk = (t >> 5) & 3;           // 64/16
    const int h    = (t >> 7) % H_;
    const int b    = t / (32 * 4 * H_);

    const f16* a_base  = Vt + ((size_t)h * E_ + eblk * 16) * D_;
    const f16* bt_base = xh + ((size_t)b * N_ + mblk * 16) * D_;

    v8f acc = {};
    for (int k0 = 0; k0 < D_; k0 += 32) {
        v16h a  = frag_a(a_base + k0, D_, lane);
        v16h bb = frag_bt(bt_base + k0, D_, lane);
        acc = wmma16(a, bb, acc);
    }
    f16* outp = xvt + ((size_t)b * H_ + h) * (size_t)(E_ * N_);
#pragma unroll
    for (int r = 0; r < 8; ++r) {
        int e = eblk * 16 + 8 * g + r;
        outp[(size_t)e * N_ + mblk * 16 + nn] = (f16)acc[r];
    }
}

// ---- Fused attention: one workgroup (8 waves) per (b, h, 32-query block) ---
__global__ __launch_bounds__(256)
void attn_kernel(const f16* __restrict__ xh, const f16* __restrict__ Qt,
                 const f16* __restrict__ xvt, float* __restrict__ out) {
    __shared__ __align__(16) f16   x_a[MT_ * D_];    // 48 KB query tile; reused as P f16[32][512]
    __shared__ __align__(16) f16   q_lds[MT_ * D_];  // 48 KB q strip
    __shared__ float scr[MT_ * 128];                 // 16 KB softmax scratch
    __shared__ float rowmax[MT_];
    __shared__ float rowinv[MT_];

    const int tid  = threadIdx.x;
    const int wave = tid >> 5;
    const int lane = tid & 31;
    const int g    = lane >> 4;
    const int nn   = lane & 15;

    const int idx  = blockIdx.x;
    const int mblk = idx & 15;                // 512/32 query blocks
    const int h    = (idx >> 4) % H_;
    const int b    = idx / (16 * H_);

    const f16* xb  = xh + (size_t)b * N_ * D_;
    const f16* qh  = Qt + (size_t)h * D_ * D_;
    const f16* xvh = xvt + ((size_t)b * H_ + h) * (size_t)(E_ * N_);

    // ---- Async stage query tile (32x768 f16 = 48 KB) into LDS
    {
        const f16* src = xb + (size_t)(mblk * MT_) * D_;
        for (int i = tid; i < (MT_ * D_) / 8; i += 256)   // 3072 x b128 chunks
            async_b128(src + i * 8, &x_a[i * 8]);
        wait_async0();
    }
    __syncthreads();

    // ---- Phase 1: q[32][768] = x_a @ Q_h
    // Wave owns 6 col-blocks x both row-blocks: each Qt fragment read ONCE.
    {
        v8f acc[2][6] = {};
        for (int k0 = 0; k0 < D_; k0 += 32) {
            v16h a0 = frag_a(x_a + k0, D_, lane);
            v16h a1 = frag_a(x_a + 16 * D_ + k0, D_, lane);
#pragma unroll
            for (int j = 0; j < 6; ++j) {
                const f16* bt = qh + (size_t)((wave * 6 + j) * 16) * D_ + k0;
                v16h bb = frag_bt(bt, D_, lane);
                acc[0][j] = wmma16(a0, bb, acc[0][j]);
                acc[1][j] = wmma16(a1, bb, acc[1][j]);
            }
        }
#pragma unroll
        for (int rb = 0; rb < 2; ++rb)
#pragma unroll
            for (int j = 0; j < 6; ++j) {
                int col = (wave * 6 + j) * 16 + nn;
#pragma unroll
                for (int r = 0; r < 8; ++r)
                    q_lds[(rb * 16 + 8 * g + r) * D_ + col] = (f16)acc[rb][j][r];
            }
    }
    __syncthreads();

    // ---- Phase 2: S (kept in registers) = q @ x^T
    // Wave owns 4 key-blocks x both row-blocks: each key fragment read ONCE.
    v8f s[2][4] = {};
    for (int k0 = 0; k0 < D_; k0 += 32) {
        v16h a0 = frag_a(q_lds + k0, D_, lane);
        v16h a1 = frag_a(q_lds + 16 * D_ + k0, D_, lane);
#pragma unroll
        for (int j = 0; j < 4; ++j) {
            const f16* bt = xb + (size_t)((wave * 4 + j) * 16) * D_ + k0;
            __builtin_prefetch(bt + 32, 0, 3);     // global_prefetch_b8 next K-chunk
            v16h bb = frag_bt(bt, D_, lane);
            s[0][j] = wmma16(a0, bb, s[0][j]);
            s[1][j] = wmma16(a1, bb, s[1][j]);
        }
    }

    // ---- Softmax over 512 keys per row, scores live in accumulators
    const int slot = wave * 16 + nn;              // 128 partials per row
#pragma unroll
    for (int rb = 0; rb < 2; ++rb)
#pragma unroll
        for (int r = 0; r < 8; ++r) {
            float m = s[rb][0][r];
#pragma unroll
            for (int j = 1; j < 4; ++j) m = fmaxf(m, s[rb][j][r]);
            scr[(rb * 16 + 8 * g + r) * 128 + slot] = m;
        }
    __syncthreads();
    if (tid < MT_) {
        float m = scr[tid * 128];
        for (int i = 1; i < 128; ++i) m = fmaxf(m, scr[tid * 128 + i]);
        rowmax[tid] = m;
    }
    __syncthreads();

    f16* P = x_a;   // query tile is dead: reuse 32 KB of it for P f16[32][512]
#pragma unroll
    for (int rb = 0; rb < 2; ++rb)
#pragma unroll
        for (int r = 0; r < 8; ++r) {
            const int row = rb * 16 + 8 * g + r;
            const float mm = rowmax[row];
            float sum = 0.f;
#pragma unroll
            for (int j = 0; j < 4; ++j) {
                float e = __expf(s[rb][j][r] - mm);
                sum += e;
                P[row * N_ + (wave * 4 + j) * 16 + nn] = (f16)e;
            }
            scr[row * 128 + slot] = sum;
        }
    __syncthreads();
    if (tid < MT_) {
        float t = 0.f;
        for (int i = 0; i < 128; ++i) t += scr[tid * 128 + i];
        rowinv[tid] = 1.0f / t;
    }
    __syncthreads();

    // ---- Phase 3: O[32][64] = P @ xv ; 8 waves = 2 row-blocks x 4 e-blocks
    {
        const int rb = wave & 1, cb = wave >> 1;
        v8f acc = {};
        for (int k0 = 0; k0 < N_; k0 += 32) {
            v16h a  = frag_a(P + rb * 16 * N_ + k0, N_, lane);
            v16h bb = frag_bt(xvh + (size_t)(cb * 16) * N_ + k0, N_, lane);
            acc = wmma16(a, bb, acc);
        }
#pragma unroll
        for (int r = 0; r < 8; ++r) {
            const int row  = rb * 16 + 8 * g + r;
            const int nrow = mblk * MT_ + row;
            out[((size_t)b * N_ + nrow) * (size_t)(H_ * E_) + h * E_ + cb * 16 + nn]
                = acc[r] * rowinv[row];
        }
    }
}

// ---- Host-side launch ------------------------------------------------------
extern "C" void kernel_launch(void* const* d_in, const int* in_sizes, int n_in,
                              void* d_out, int out_size, void* d_ws, size_t ws_size,
                              hipStream_t stream) {
    const float* x = (const float*)d_in[0];   // [32][512][768]
    const float* Q = (const float*)d_in[1];   // [12][768][768]
    const float* V = (const float*)d_in[2];   // [12][768][64]
    float* out = (float*)d_out;               // [32][512][768]
    char* ws = (char*)d_ws;

    // workspace layout (f16): xh 24MB | Qt 13.5MB | Vt 1.1MB | xvt 24MB (~66MB)
    const size_t off_xh  = 0;
    const size_t off_qt  = off_xh  + (size_t)B_ * N_ * D_ * 2;
    const size_t off_vt  = off_qt  + (size_t)H_ * D_ * D_ * 2;
    const size_t off_xvt = off_vt  + (size_t)H_ * E_ * D_ * 2;
    f16* xh  = (f16*)(ws + off_xh);
    f16* Qt  = (f16*)(ws + off_qt);
    f16* Vt  = (f16*)(ws + off_vt);
    f16* xvt = (f16*)(ws + off_xvt);

    const int nx = B_ * N_ * D_;
    cvt_x_kernel <<<(nx + 255) / 256, 256, 0, stream>>>(x, xh, nx);
    cvt_qt_kernel<<<(H_ * D_ * D_ + 255) / 256, 256, 0, stream>>>(Q, Qt);
    cvt_vt_kernel<<<(H_ * D_ * E_ + 255) / 256, 256, 0, stream>>>(V, Vt);

    // 32*12*4*32 = 49152 tiles / 8 waves per block
    xv_kernel<<<6144, 256, 0, stream>>>(xh, Vt, xvt);

    // one block per (b, h, 32-row query block): 32*12*16
    attn_kernel<<<6144, 256, 0, stream>>>(xh, Qt, xvt, out);
}
